// VanillaRNN_43636867727443
// MI455X (gfx1250) — matrix operand order repaired and checked by reading, not compiled, not run on hardware
//
#include <hip/hip_runtime.h>
#include <hip/hip_bf16.h>
#include <math.h>

// ---------------- problem constants (from reference) ----------------
#define SEQ_T    256
#define HDIM     1024
#define BATCH    4096
#define NCLS     10

// ---------------- tiling ----------------
#define BT       32                 // batch columns per workgroup (128 WGs total)
#define NT       (BT / 16)          // N-tiles per workgroup (2)
#define HP       (HDIM + 8)         // padded LDS column stride (bank-conflict avoidance)
#define NTHREADS 256                // 8 waves (wave32)
#define NWAVES   8
#define KTILES   (HDIM / 32)        // 32 k-steps

typedef __attribute__((ext_vector_type(16))) __bf16 v16bf;
typedef __attribute__((ext_vector_type(8)))  __bf16 v8bf;
typedef __attribute__((ext_vector_type(8)))  float  v8f;

// Branch-free tanh: prefer gfx1250's V_TANH_F32; fall back to
// tanh(x) = 1 - 2/(exp2(2*log2(e)*x)+1) (saturates correctly, no EXEC div).
__device__ __forceinline__ float fast_tanh(float v) {
#if __has_builtin(__builtin_amdgcn_tanhf)
    return __builtin_amdgcn_tanhf(v);
#else
    const float e = __builtin_amdgcn_exp2f(v * 2.885390081777926815f); // 2*log2(e)
    const float r = __builtin_amdgcn_rcpf(e + 1.0f);
    return 1.0f - 2.0f * r;
#endif
}

// ---------------- prologue: W_hh fp32 -> bf16 in workspace ----------------
__global__ __launch_bounds__(256)
void whh_to_bf16(const float* __restrict__ W, __bf16* __restrict__ Wb, int n) {
    int i = blockIdx.x * blockDim.x + threadIdx.x;
    int stride = gridDim.x * blockDim.x;
    for (; i < n; i += stride) Wb[i] = (__bf16)W[i];
}

// ---------------- main RNN kernel ----------------
// One workgroup = one 32-column batch tile, full time loop, h in LDS (bf16,
// double buffered, column-major with padded stride). 132 KB LDS -> two WGs
// can co-reside on one 320 KB WGP, sharing streamed W_hh lines in WGP$.
__global__ __launch_bounds__(NTHREADS)
void rnn_wmma_kernel(const float* __restrict__ x,       // [BATCH, SEQ_T]
                     const float* __restrict__ W_hx,    // [HDIM]
                     const __bf16* __restrict__ Wb,     // [HDIM, HDIM] bf16 row-major
                     const float* __restrict__ W_ph,    // [NCLS, HDIM]
                     const float* __restrict__ bias_h,  // [HDIM]
                     const float* __restrict__ bias_p,  // [NCLS]
                     float* __restrict__ out)           // [BATCH, NCLS]
{
    extern __shared__ char smem[];
    __bf16* buf0 = (__bf16*)smem;          // h buffers, column-major [BT][HP]
    __bf16* buf1 = buf0 + BT * HP;

    const int tid   = threadIdx.x;
    const int wave  = tid >> 5;
    const int lane  = tid & 31;
    const int laneM = lane & 15;           // row-within-tile / column-within-tile
    const int half  = lane >> 4;           // 0 or 1
    const int b0    = blockIdx.x * BT;     // global batch offset of this WG

    // h0 = 0
    for (int idx = tid; idx < BT * HP; idx += NTHREADS) buf0[idx] = (__bf16)0.f;
    __syncthreads();

    int cur = 0;
    #pragma unroll 1
    for (int t = 0; t < SEQ_T; ++t) {
        const __bf16* hcur  = cur ? buf1 : buf0;
        __bf16*       hnext = cur ? buf0 : buf1;

        // 8 waves x 2 iterations x 4 M-tiles = 64 M-tiles (HDIM/16)
        #pragma unroll 1
        for (int it = 0; it < 2; ++it) {
            const int mt0 = (wave * 2 + it) * 4;     // first of 4 M-tiles

            v8f acc[4][NT] = {};                      // C tiles, fp32

            // A row base pointers (W_hh bf16, row-major): A-frag layout for
            // 16x32 bf16: lane<16 gets K {k0..k0+7, k0+16..k0+23},
            // lane>=16 gets K {k0+8..k0+15, k0+24..k0+31}.
            const __bf16* arow[4];
            #pragma unroll
            for (int m = 0; m < 4; ++m)
                arow[m] = Wb + (size_t)((mt0 + m) * 16 + laneM) * HDIM + half * 8;

            // B column base pointers (h in LDS, column-major): B-frag layout
            // for 32x16 bf16: lane<16 -> K {k0..k0+7, k0+16..k0+23} of column laneM,
            // lane>=16 -> K {k0+8..k0+15, k0+24..k0+31} of column laneM.
            const __bf16* bcol[NT];
            #pragma unroll
            for (int n = 0; n < NT; ++n)
                bcol[n] = hcur + (size_t)(n * 16 + laneM) * HP + half * 8;

            // Unroll by 4: intra-body k-offsets (0..224B) fold into the load
            // immediates (24-bit global IOFFSET / 16-bit DS offset), so the
            // pointer-advance VALU (and its WMMA->VALU hazard NOPs) is paid
            // once per 32 WMMAs instead of once per 8.
            #pragma unroll 4
            for (int kt = 0; kt < KTILES; ++kt) {
                const int k0 = kt * 32;

                v16bf Bf[NT];
                #pragma unroll
                for (int n = 0; n < NT; ++n) {
                    v8bf lo = *(const v8bf*)(bcol[n] + k0);
                    v8bf hi = *(const v8bf*)(bcol[n] + k0 + 16);
                    Bf[n] = __builtin_shufflevector(lo, hi,
                        0,1,2,3,4,5,6,7,8,9,10,11,12,13,14,15);
                }

                v16bf Af[4];
                #pragma unroll
                for (int m = 0; m < 4; ++m) {
                    v8bf a0 = *(const v8bf*)(arow[m] + k0);
                    v8bf a1 = *(const v8bf*)(arow[m] + k0 + 16);
                    Af[m] = __builtin_shufflevector(a0, a1,
                        0,1,2,3,4,5,6,7,8,9,10,11,12,13,14,15);
                }

                #pragma unroll
                for (int m = 0; m < 4; ++m)
                    #pragma unroll
                    for (int n = 0; n < NT; ++n)
                        acc[m][n] = __builtin_amdgcn_wmma_f32_16x16x32_bf16(
                            false, Af[m], false, Bf[n],
                            (short)0, acc[m][n], false, false);
            }

            // Epilogue: h_new = tanh(acc + W_hx*x_t + bias_h), pack to bf16,
            // store 8 consecutive rows (one b128) per lane into hnext.
            float xv[NT];
            #pragma unroll
            for (int n = 0; n < NT; ++n)
                xv[n] = x[(size_t)(b0 + n * 16 + laneM) * SEQ_T + t];

            #pragma unroll
            for (int m = 0; m < 4; ++m) {
                const int i0 = (mt0 + m) * 16 + half * 8;   // first of 8 rows
                float whx[8], bh[8];
                #pragma unroll
                for (int r = 0; r < 8; ++r) {
                    whx[r] = W_hx[i0 + r];
                    bh[r]  = bias_h[i0 + r];
                }
                #pragma unroll
                for (int n = 0; n < NT; ++n) {
                    const int col = n * 16 + laneM;         // column within WG
                    unsigned int packed[4];
                    #pragma unroll
                    for (int r = 0; r < 4; ++r) {
                        float v0 = fast_tanh(acc[m][n][2*r]   + whx[2*r]   * xv[n] + bh[2*r]);
                        float v1 = fast_tanh(acc[m][n][2*r+1] + whx[2*r+1] * xv[n] + bh[2*r+1]);
                        union { __bf16 h[2]; unsigned int u; } cvt;
                        cvt.h[0] = (__bf16)v0;
                        cvt.h[1] = (__bf16)v1;
                        packed[r] = cvt.u;
                    }
                    uint4 uv;
                    uv.x = packed[0]; uv.y = packed[1];
                    uv.z = packed[2]; uv.w = packed[3];
                    *(uint4*)(void*)(hnext + (size_t)col * HP + i0) = uv;
                }
            }
        }
        __syncthreads();
        cur ^= 1;
    }

    // ---------------- final projection: out[b,c] = W_ph @ h_final + bias_p ----
    // SEQ_T is even -> final h is in buf0; reuse buf1 as fp32 partial scratch.
    const __bf16* hf = buf0;
    float* partial = (float*)buf1;          // [BT][NCLS][8] partials

    const int col  = tid & (BT - 1);
    const int part = tid / BT;              // 0..7 -> 128-row slice each
    const __bf16* hc = hf + (size_t)col * HP;

    float s[NCLS];
    #pragma unroll
    for (int c = 0; c < NCLS; ++c) s[c] = 0.f;

    const int j0 = part * (HDIM / 8);
    #pragma unroll 1
    for (int j = j0; j < j0 + HDIM / 8; ++j) {
        const float hv = (float)hc[j];
        #pragma unroll
        for (int c = 0; c < NCLS; ++c)
            s[c] += W_ph[c * HDIM + j] * hv;
    }
    #pragma unroll
    for (int c = 0; c < NCLS; ++c)
        partial[(col * NCLS + c) * 8 + part] = s[c];
    __syncthreads();

    if (part == 0) {
        #pragma unroll
        for (int c = 0; c < NCLS; ++c) {
            const float* pp = &partial[(col * NCLS + c) * 8];
            float acc_c = bias_p[c];
            #pragma unroll
            for (int p = 0; p < 8; ++p) acc_c += pp[p];
            out[(size_t)(b0 + col) * NCLS + c] = acc_c;
        }
    }
}

extern "C" void kernel_launch(void* const* d_in, const int* in_sizes, int n_in,
                              void* d_out, int out_size, void* d_ws, size_t ws_size,
                              hipStream_t stream) {
    (void)in_sizes; (void)n_in; (void)out_size; (void)ws_size;

    const float* x      = (const float*)d_in[0];  // [4096,256]
    const float* W_hx   = (const float*)d_in[1];  // [1024,1]
    const float* W_hh   = (const float*)d_in[2];  // [1024,1024]
    const float* W_ph   = (const float*)d_in[3];  // [10,1024]
    const float* bias_h = (const float*)d_in[4];  // [1024,1]
    const float* bias_p = (const float*)d_in[5];  // [10,1]

    __bf16* Wb = (__bf16*)d_ws;                   // 2 MB bf16 copy of W_hh

    whh_to_bf16<<<1024, 256, 0, stream>>>(W_hh, Wb, HDIM * HDIM);

    const size_t shmem = 2 * (size_t)BT * HP * sizeof(__bf16);  // ~132 KB -> 2 WGs/WGP
    rnn_wmma_kernel<<<BATCH / BT, NTHREADS, shmem, stream>>>(
        x, W_hx, Wb, W_ph, bias_h, bias_p, (float*)d_out);
}
